// IGNNLayer_11020886082098
// MI455X (gfx1250) — compile-verified, matrix operand order checked by reading, not run on hardware
//
#include <hip/hip_runtime.h>
#include <hip/hip_bf16.h>
#include <math.h>

// ---------------------------------------------------------------------------
// IGNN layer for MI455X (gfx1250): bf16 WMMA projections + fused flash-style
// masked attention + residual + LayerNorm.
//  - V projection stores V^T so attention B-fragments are contiguous b128s.
//  - adj tiles staged with global_load_async_to_lds_b128 (ASYNCcnt), double-
//    buffered LDS, one barrier per j-step.
//  - GEMM K-loop software-pipelined (register double buffering).
// ---------------------------------------------------------------------------

typedef __attribute__((ext_vector_type(16))) __bf16 v16bf;
typedef __attribute__((ext_vector_type(8)))  __bf16 v8bf;
typedef __attribute__((ext_vector_type(8)))  float  v8f;

#define NN    4096
#define DIMM  256
#define HEADS 4
#define DKK   64   // DIMM / HEADS

// 16-bit WMMA operand fragment from two contiguous 8-element (16B) runs.
// Layout (ISA 7.12.2): lane&15 -> row, lane>>4 -> K-half; e0..7 cover
// K = kh*8.., e8..15 cover K = 16+kh*8.. — both contiguous.
__device__ __forceinline__ v16bf frag_from(const __bf16* p0, const __bf16* p1) {
  v8bf lo = *(const v8bf*)p0;
  v8bf hi = *(const v8bf*)p1;
  v16bf f;
#pragma unroll
  for (int e = 0; e < 8; ++e) { f[e] = lo[e]; f[e + 8] = hi[e]; }
  return f;
}

// LDS byte offset of a generic pointer to __shared__ (low 32 bits of the
// shared-aperture generic address are the LDS offset).
__device__ __forceinline__ unsigned lds_off(const void* p) {
  return (unsigned)(uintptr_t)p;
}

// ---------------------------------------------------------------------------
// fp32 -> bf16 convert (grid-stride)
// ---------------------------------------------------------------------------
__global__ void f32_to_bf16_kernel(const float* __restrict__ src,
                                   __bf16* __restrict__ dst, int n) {
  for (int i = blockIdx.x * blockDim.x + threadIdx.x; i < n;
       i += gridDim.x * blockDim.x)
    dst[i] = (__bf16)src[i];
}

// ---------------------------------------------------------------------------
// C = A[M,K](bf16) * B[Nc,K]^T(bf16) + bias.
// mode: 0 -> fp32 row-major, 1 -> bf16 row-major, 2 -> bf16 TRANSPOSED
// (C[col*M + row], used to emit V^T for the attention kernel).
// Block: 128 threads = 4 waves; block tile 128x64; wave w -> rows w*32..+32.
// K-loop software-pipelined with register double buffering.
// ---------------------------------------------------------------------------
__global__ __launch_bounds__(128) void gemm_bias_kernel(
    const __bf16* __restrict__ A, const __bf16* __restrict__ B,
    const float* __restrict__ bias, void* __restrict__ C,
    int M, int Nc, int K, int mode) {
  const int lane = threadIdx.x & 31;
  const int wave = threadIdx.x >> 5;
  const int l15  = lane & 15;
  const int kh   = (lane >> 4) & 1;
  const int r0   = blockIdx.x * 128 + wave * 32;
  const int c0   = blockIdx.y * 64;

  v8f acc[2][4] = {};
  v16bf af[2][2], bfr[2][4];
  int cur = 0;

  // prologue: fragments for k0 = 0
#pragma unroll
  for (int u = 0; u < 2; ++u) {
    const __bf16* ap = A + (size_t)(r0 + u * 16 + l15) * K + kh * 8;
    af[cur][u] = frag_from(ap, ap + 16);
  }
#pragma unroll
  for (int t = 0; t < 4; ++t) {
    const __bf16* bp = B + (size_t)(c0 + t * 16 + l15) * K + kh * 8;
    bfr[cur][t] = frag_from(bp, bp + 16);
  }

  for (int k0 = 0; k0 < K; k0 += 32) {
    const int nxt = cur ^ 1;
    if (k0 + 32 < K) {  // issue next tile's loads before this tile's WMMAs
#pragma unroll
      for (int u = 0; u < 2; ++u) {
        const __bf16* ap = A + (size_t)(r0 + u * 16 + l15) * K + k0 + 32 + kh * 8;
        af[nxt][u] = frag_from(ap, ap + 16);
      }
#pragma unroll
      for (int t = 0; t < 4; ++t) {
        const __bf16* bp = B + (size_t)(c0 + t * 16 + l15) * K + k0 + 32 + kh * 8;
        bfr[nxt][t] = frag_from(bp, bp + 16);
      }
    }
#pragma unroll
    for (int u = 0; u < 2; ++u)
#pragma unroll
      for (int t = 0; t < 4; ++t)
        acc[u][t] = __builtin_amdgcn_wmma_f32_16x16x32_bf16(
            false, af[cur][u], false, bfr[cur][t], (short)0, acc[u][t],
            false, false);
    cur = nxt;
  }

#pragma unroll
  for (int u = 0; u < 2; ++u) {
#pragma unroll
    for (int t = 0; t < 4; ++t) {
      const int col = c0 + t * 16 + l15;
      const float b = bias[col];
#pragma unroll
      for (int v = 0; v < 8; ++v) {
        const int row = r0 + u * 16 + v + 8 * kh;
        const float val = acc[u][t][v] + b;
        if (mode == 0)      ((float*)C)[(size_t)row * Nc + col] = val;
        else if (mode == 1) ((__bf16*)C)[(size_t)row * Nc + col] = (__bf16)val;
        else                ((__bf16*)C)[(size_t)col * M + row] = (__bf16)val;
      }
    }
  }
}

// ---------------------------------------------------------------------------
// Fused masked flash attention.
// Block = 128 threads = 4 waves; wave h owns head h for a 16-row i-block.
// adj tiles: async DMA to double-buffered LDS (ASYNCcnt), 1 barrier/step.
// V supplied TRANSPOSED [DIM, N] so PV B-fragments are contiguous b128 loads.
// ---------------------------------------------------------------------------
__global__ __launch_bounds__(128) void attn_kernel(
    const __bf16* __restrict__ Qm, const __bf16* __restrict__ Km,
    const __bf16* __restrict__ Vt, const float* __restrict__ adj,
    __bf16* __restrict__ Out) {
  __shared__ __align__(16) float  adjsh[2][16][32];
  __shared__ __align__(16) __bf16 psh[HEADS][16][32];

  const int tid  = threadIdx.x;
  const int lane = tid & 31;
  const int h    = tid >> 5;
  const int l15  = lane & 15;
  const int kh   = (lane >> 4) & 1;
  const int i0   = blockIdx.x * 16;

  // Resident Q fragments (dk = 64 -> two K=32 fragments)
  v16bf qf[2];
#pragma unroll
  for (int ks = 0; ks < 2; ++ks) {
    const __bf16* p = Qm + (size_t)(i0 + l15) * DIMM + h * DKK + ks * 32 + kh * 8;
    qf[ks] = frag_from(p, p + 16);
  }

  v8f acc[4] = {};
  float mrow[8], lrow[8];
#pragma unroll
  for (int v = 0; v < 8; ++v) { mrow[v] = -3.0e38f; lrow[v] = 0.0f; }

  // adj staging: one async b128 (4 floats) per thread covers 16x32 tile
  const int arow = tid >> 3;
  const int acol = (tid & 7) * 4;

  // prestage j-block 0 into buffer 0
  {
    const unsigned dst = lds_off(&adjsh[0][arow][acol]);
    const float* src = adj + (size_t)(i0 + arow) * NN + acol;
    asm volatile("global_load_async_to_lds_b128 %0, %1, off"
                 :: "v"(dst), "v"(src) : "memory");
  }

  const int NB = NN / 32;
  for (int jb = 0; jb < NB; ++jb) {
    const int j0 = jb * 32;
    asm volatile("s_wait_asynccnt 0x0" ::: "memory");
    __syncthreads();
    if (jb + 1 < NB) {  // DMA next tile while computing this one
      const unsigned dst = lds_off(&adjsh[(jb + 1) & 1][arow][acol]);
      const float* src = adj + (size_t)(i0 + arow) * NN + j0 + 32 + acol;
      asm volatile("global_load_async_to_lds_b128 %0, %1, off"
                   :: "v"(dst), "v"(src) : "memory");
    }
    const int buf = jb & 1;

    // ---- scores: two 16x16 tiles over this 32-wide j block, K-dim 64 ----
    v8f s[2] = {};
#pragma unroll
    for (int T = 0; T < 2; ++T) {
#pragma unroll
      for (int ks = 0; ks < 2; ++ks) {
        const __bf16* p =
            Km + (size_t)(j0 + T * 16 + l15) * DIMM + h * DKK + ks * 32 + kh * 8;
        v16bf kf = frag_from(p, p + 16);
        s[T] = __builtin_amdgcn_wmma_f32_16x16x32_bf16(
            false, qf[ks], false, kf, (short)0, s[T], false, false);
      }
      if (j0 + 32 < NN)
        __builtin_prefetch(
            Km + (size_t)(j0 + 32 + T * 16 + l15) * DIMM + h * DKK, 0, 1);
    }

    // ---- multiplicative mask + scale, online softmax over 32 columns ----
    float ms[2][8], mnew[8];
#pragma unroll
    for (int v = 0; v < 8; ++v) {
      const int r = v + 8 * kh;
      ms[0][v] = s[0][v] * 0.125f * adjsh[buf][r][l15];
      ms[1][v] = s[1][v] * 0.125f * adjsh[buf][r][16 + l15];
      mnew[v]  = fmaxf(mrow[v], fmaxf(ms[0][v], ms[1][v]));
    }
#pragma unroll
    for (int v = 0; v < 8; ++v) {
#pragma unroll
      for (int d = 1; d < 16; d <<= 1)  // reduce within 16-lane half
        mnew[v] = fmaxf(mnew[v], __shfl_xor(mnew[v], d, 32));
    }
    float fac[8], rsum[8];
#pragma unroll
    for (int v = 0; v < 8; ++v) {
      fac[v] = __expf(mrow[v] - mnew[v]);
      const float p0 = __expf(ms[0][v] - mnew[v]);
      const float p1 = __expf(ms[1][v] - mnew[v]);
      const int r = v + 8 * kh;
      psh[h][r][l15]      = (__bf16)p0;  // C-layout -> LDS row-major 16x32
      psh[h][r][16 + l15] = (__bf16)p1;
      rsum[v] = p0 + p1;
      mrow[v] = mnew[v];
    }
#pragma unroll
    for (int v = 0; v < 8; ++v) {
#pragma unroll
      for (int d = 1; d < 16; d <<= 1)
        rsum[v] += __shfl_xor(rsum[v], d, 32);
      lrow[v] = lrow[v] * fac[v] + rsum[v];
    }

    // ---- P: per-wave LDS round-trip re-layout C-frag -> A-frag ----
    const __bf16* pp = &psh[h][l15][kh * 8];
    v16bf pf = frag_from(pp, pp + 16);

    // ---- acc = acc*fac + P @ V  (V^T: contiguous j-runs per d-column) ----
#pragma unroll
    for (int t = 0; t < 4; ++t) {
#pragma unroll
      for (int v = 0; v < 8; ++v) acc[t][v] *= fac[v];
      const __bf16* vp =
          Vt + (size_t)(h * DKK + t * 16 + l15) * NN + j0 + kh * 8;
      v16bf vf = frag_from(vp, vp + 16);
      acc[t] = __builtin_amdgcn_wmma_f32_16x16x32_bf16(
          false, pf, false, vf, (short)0, acc[t], false, false);
    }
  }

  // ---- epilogue ----
#pragma unroll
  for (int v = 0; v < 8; ++v) {
    const float inv = 1.0f / lrow[v];
    const int row = i0 + v + 8 * kh;
#pragma unroll
    for (int t = 0; t < 4; ++t)
      Out[(size_t)row * DIMM + h * DKK + t * 16 + l15] =
          (__bf16)(acc[t][v] * inv);
  }
}

// ---------------------------------------------------------------------------
// out = LayerNorm(oproj + x) * gamma + beta   (one row per 256-thread block)
// ---------------------------------------------------------------------------
__global__ __launch_bounds__(256) void ln_kernel(
    const float* __restrict__ oproj, const float* __restrict__ x,
    const float* __restrict__ gamma, const float* __restrict__ beta,
    float* __restrict__ out) {
  __shared__ float red[256];
  const int row = blockIdx.x, t = threadIdx.x;
  const float hv = oproj[(size_t)row * DIMM + t] + x[(size_t)row * DIMM + t];
  red[t] = hv;
  __syncthreads();
  for (int s = 128; s > 0; s >>= 1) {
    if (t < s) red[t] += red[t + s];
    __syncthreads();
  }
  const float mu = red[0] * (1.0f / DIMM);
  __syncthreads();
  const float d = hv - mu;
  red[t] = d * d;
  __syncthreads();
  for (int s = 128; s > 0; s >>= 1) {
    if (t < s) red[t] += red[t + s];
    __syncthreads();
  }
  const float var = red[0] * (1.0f / DIMM);
  out[(size_t)row * DIMM + t] = d * rsqrtf(var + 1e-5f) * gamma[t] + beta[t];
}

// ---------------------------------------------------------------------------
extern "C" void kernel_launch(void* const* d_in, const int* in_sizes, int n_in,
                              void* d_out, int out_size, void* d_ws,
                              size_t ws_size, hipStream_t stream) {
  const float* x     = (const float*)d_in[0];
  const float* adj   = (const float*)d_in[1];
  const float* Wq    = (const float*)d_in[2];
  const float* bq    = (const float*)d_in[3];
  const float* Wk    = (const float*)d_in[4];
  const float* bk    = (const float*)d_in[5];
  const float* Wv    = (const float*)d_in[6];
  const float* bv    = (const float*)d_in[7];
  const float* Wo    = (const float*)d_in[8];
  const float* bo    = (const float*)d_in[9];
  const float* gamma = (const float*)d_in[10];
  const float* beta  = (const float*)d_in[11];
  float* out = (float*)d_out;

  char* ws = (char*)d_ws;
  size_t o = 0;
  __bf16* xbf  = (__bf16*)(ws + o); o += (size_t)NN * DIMM * 2;
  __bf16* wqbf = (__bf16*)(ws + o); o += (size_t)DIMM * DIMM * 2;
  __bf16* wkbf = (__bf16*)(ws + o); o += (size_t)DIMM * DIMM * 2;
  __bf16* wvbf = (__bf16*)(ws + o); o += (size_t)DIMM * DIMM * 2;
  __bf16* wobf = (__bf16*)(ws + o); o += (size_t)DIMM * DIMM * 2;
  __bf16* qbf  = (__bf16*)(ws + o); o += (size_t)NN * DIMM * 2;
  __bf16* kbf  = (__bf16*)(ws + o); o += (size_t)NN * DIMM * 2;
  __bf16* vtbf = (__bf16*)(ws + o); o += (size_t)NN * DIMM * 2;   // V^T
  __bf16* abf  = (__bf16*)(ws + o); o += (size_t)NN * DIMM * 2;   // attn out
  float*  oprj = (float*)(ws + o);  o += (size_t)NN * DIMM * 4;

  // 1) fp32 -> bf16 conversions
  f32_to_bf16_kernel<<<512, 256, 0, stream>>>(x, xbf, NN * DIMM);
  f32_to_bf16_kernel<<<128, 256, 0, stream>>>(Wq, wqbf, DIMM * DIMM);
  f32_to_bf16_kernel<<<128, 256, 0, stream>>>(Wk, wkbf, DIMM * DIMM);
  f32_to_bf16_kernel<<<128, 256, 0, stream>>>(Wv, wvbf, DIMM * DIMM);
  f32_to_bf16_kernel<<<128, 256, 0, stream>>>(Wo, wobf, DIMM * DIMM);

  // 2) projections: Q,K row-major bf16; V transposed bf16
  dim3 ggrid(NN / 128, DIMM / 64);
  gemm_bias_kernel<<<ggrid, 128, 0, stream>>>(xbf, wqbf, bq, qbf,
                                              NN, DIMM, DIMM, 1);
  gemm_bias_kernel<<<ggrid, 128, 0, stream>>>(xbf, wkbf, bk, kbf,
                                              NN, DIMM, DIMM, 1);
  gemm_bias_kernel<<<ggrid, 128, 0, stream>>>(xbf, wvbf, bv, vtbf,
                                              NN, DIMM, DIMM, 2);

  // 3) fused masked flash attention
  attn_kernel<<<NN / 16, 128, 0, stream>>>(qbf, kbf, vtbf, adj, abf);

  // 4) output projection (fp32)
  gemm_bias_kernel<<<ggrid, 128, 0, stream>>>(abf, wobf, bo, oprj,
                                              NN, DIMM, DIMM, 0);

  // 5) residual + LayerNorm
  ln_kernel<<<NN, 256, 0, stream>>>(oprj, x, gamma, beta, out);
}
